// PatchRnn_backbone_40785009443528
// MI455X (gfx1250) — compile-verified, matrix operand order checked by reading, not run on hardware
//
#include <hip/hip_runtime.h>
#include <hip/hip_bf16.h>
#include <stdint.h>

// ---------------- problem constants ----------------
#define BB     64
#define NVV    21
#define LL     2048
#define PATCH  16
#define STRIDE 8
#define PNUM   255
#define DD     128
#define HH     256
#define PREDN  96
#define BN     (BB*NVV)          // 1344
#define NROWS  (BN*PNUM)         // 342720
#define KTOT   (DD*PNUM)         // 32640
#define EPSLN  1e-5f
#define EMASCALE 0.70710678118654752440f  // 1/sqrt(2)

typedef __bf16 bf16_t;
typedef __attribute__((ext_vector_type(16))) __bf16 bf16x16;
typedef __attribute__((ext_vector_type(8)))  float  f32x8;

struct U4x2 { uint4 lo, hi; };

static __device__ inline bf16_t f2bf(float f) { return (bf16_t)f; }
static __device__ inline float  bf2f(bf16_t b){ return (float)b; }
static __device__ inline float  sigmoidf_(float v){ return 1.0f/(1.0f+__expf(-v)); }

// Load a 16x32 bf16 fragment (A or B^T) from row-major LDS/mem.
// Lane layout per CDNA5 ISA 7.12.2 (16-bit A 16x32):
//   lane<16 : elems 0..7 -> K=kbase+0..7,  elems 8..15 -> K=kbase+16..23
//   lane>=16: elems 0..7 -> K=kbase+8..15, elems 8..15 -> K=kbase+24..31
static __device__ inline bf16x16 load_frag(const bf16_t* base, int row0, int ld,
                                           int kbase, int lane) {
  int m = lane & 15, half = lane >> 4;
  const bf16_t* p = base + (size_t)(row0 + m) * ld + kbase + half * 8;
  U4x2 t;
  t.lo = *(const uint4*)p;
  t.hi = *(const uint4*)(p + 16);
  return __builtin_bit_cast(bf16x16, t);
}

static __device__ inline f32x8 wmma_bf16(bf16x16 a, bf16x16 b, f32x8 c) {
  return __builtin_amdgcn_wmma_f32_16x16x32_bf16(false, a, false, b,
                                                 (short)0, c, false, false);
}

// ---------------- kernel 0: head_w f32 -> bf16 ----------------
__global__ __launch_bounds__(256) void cvt_kernel(const float* __restrict__ w,
                                                  bf16_t* __restrict__ o, int n) {
  int i = blockIdx.x * 256 + threadIdx.x;
  if (i < n) o[i] = f2bf(w[i]);
}

// ---------------- kernel 1: patch + feature-projection LEMblock ----------------
// out z0: [NROWS][128] bf16
__global__ __launch_bounds__(256) void fp_kernel(
    const float* __restrict__ x,
    const float* __restrict__ w1, const float* __restrict__ b1,
    const float* __restrict__ w2, const float* __restrict__ b2,
    const float* __restrict__ wr, const float* __restrict__ br,
    const float* __restrict__ g,  const float* __restrict__ be,
    bf16_t* __restrict__ zout) {
  __shared__ __align__(16) char smem[53248];
  bf16_t* xA = (bf16_t*)(smem);          // [64][32]  (K padded 16->32)   4KB
  bf16_t* wT = (bf16_t*)(smem + 4096);   // w1T[256][32]/wrT[128][32]/w2T[128][64] 16KB
  bf16_t* hS = (bf16_t*)(smem + 20480);  // [64][256] bf16               32KB
  float*  yS = (float*)(smem + 20480);   // overlay: [64][128] f32       32KB

  const int tid = threadIdx.x, lane = tid & 31, wave = tid >> 5;
  const int row0 = blockIdx.x * 64;

  // stage patches (zero-padded K)
  for (int i = tid; i < 64 * 32; i += 256) {
    int r = i >> 5, k = i & 31;
    int gr = row0 + r, bn = gr / PNUM, p = gr - bn * PNUM;
    float v = (k < PATCH) ? x[(size_t)bn * LL + p * STRIDE + k] : 0.0f;
    xA[r * 32 + k] = f2bf(v);
  }
  // stage w1^T [H][32]
  for (int i = tid; i < HH * 32; i += 256) {
    int n = i >> 5, k = i & 31;
    wT[n * 32 + k] = f2bf((k < PATCH) ? w1[k * HH + n] : 0.0f);
  }
  __syncthreads();

  // GEMM1: h = relu(x@w1+b1)  (64x256, 64 tiles / 8 waves)
#pragma unroll
  for (int i = 0; i < 8; ++i) {
    int idx = i * 8 + wave, mt = idx >> 4, nt = idx & 15;
    f32x8 acc = {};
    acc = wmma_bf16(load_frag(xA, mt * 16, 32, 0, lane),
                    load_frag(wT, nt * 16, 32, 0, lane), acc);
    int n = lane & 15, half = lane >> 4, col = nt * 16 + n;
    float bias = b1[col];
#pragma unroll
    for (int r = 0; r < 8; ++r) {
      float v = acc[r] + bias;
      hS[(mt * 16 + half * 8 + r) * HH + col] = f2bf(v > 0.f ? v : 0.f);
    }
  }
  __syncthreads();

  // stage wr^T [128][32]
  for (int i = tid; i < DD * 32; i += 256) {
    int n = i >> 5, k = i & 31;
    wT[n * 32 + k] = f2bf((k < PATCH) ? wr[k * DD + n] : 0.0f);
  }
  __syncthreads();

  // residual GEMM into persistent accumulators (32 tiles / 8 waves)
  f32x8 racc[4];
#pragma unroll
  for (int i = 0; i < 4; ++i) {
    int idx = i * 8 + wave, mt = idx >> 3, nt = idx & 7;
    f32x8 acc = {};
    racc[i] = wmma_bf16(load_frag(xA, mt * 16, 32, 0, lane),
                        load_frag(wT, nt * 16, 32, 0, lane), acc);
  }

  // GEMM2: y += h @ w2, K=256 in 4 chunks of 64
  for (int kc = 0; kc < HH; kc += 64) {
    __syncthreads();
    for (int i = tid; i < DD * 64; i += 256) {
      int n = i >> 6, k = i & 63;
      wT[n * 64 + k] = f2bf(w2[(kc + k) * DD + n]);
    }
    __syncthreads();
#pragma unroll
    for (int i = 0; i < 4; ++i) {
      int idx = i * 8 + wave, mt = idx >> 3, nt = idx & 7;
#pragma unroll
      for (int kk = 0; kk < 64; kk += 32)
        racc[i] = wmma_bf16(load_frag(hS, mt * 16, HH, kc + kk, lane),
                            load_frag(wT, nt * 16, 64, kk, lane), racc[i]);
    }
  }
  __syncthreads();

  // y -> LDS (overlay h), add biases
#pragma unroll
  for (int i = 0; i < 4; ++i) {
    int idx = i * 8 + wave, mt = idx >> 3, nt = idx & 7;
    int n = lane & 15, half = lane >> 4, col = nt * 16 + n;
    float bias = b2[col] + br[col];
#pragma unroll
    for (int r = 0; r < 8; ++r)
      yS[(mt * 16 + half * 8 + r) * DD + col] = racc[i][r] + bias;
  }
  __syncthreads();

  // layernorm + store bf16
  if (tid < 64) {
    const float* yr = yS + tid * DD;
    float mu = 0.f;
    for (int c = 0; c < DD; ++c) mu += yr[c];
    mu *= (1.0f / DD);
    float var = 0.f;
    for (int c = 0; c < DD; ++c) { float d = yr[c] - mu; var += d * d; }
    float rs = rsqrtf(var * (1.0f / DD) + EPSLN);
    size_t base = (size_t)(row0 + tid) * DD;
    for (int c = 0; c < DD; ++c)
      zout[base + c] = f2bf(g[c] * (yr[c] - mu) * rs + be[c]);
  }
}

// ---------------- kernel 2: encoder LEMblock (D->D->D) ----------------
__global__ __launch_bounds__(256) void enc_kernel(
    const bf16_t* __restrict__ zin,
    const float* __restrict__ w1, const float* __restrict__ b1,
    const float* __restrict__ w2, const float* __restrict__ b2,
    const float* __restrict__ wr, const float* __restrict__ br,
    const float* __restrict__ g,  const float* __restrict__ be,
    bf16_t* __restrict__ zout) {
  __shared__ __align__(16) char smem[65536];
  bf16_t* xA = (bf16_t*)(smem);           // [64][128] 16KB
  bf16_t* hS = (bf16_t*)(smem + 16384);   // [64][128] 16KB
  bf16_t* wT = (bf16_t*)(smem + 32768);   // [128][128] 32KB
  float*  yS = (float*)(smem + 32768);    // overlay after GEMMs

  const int tid = threadIdx.x, lane = tid & 31, wave = tid >> 5;
  const int row0 = blockIdx.x * 64;

  {
    const uint4* src = (const uint4*)(zin + (size_t)row0 * DD);
    uint4* dst = (uint4*)xA;
    for (int i = tid; i < 1024; i += 256) dst[i] = src[i];
  }
  for (int i = tid; i < DD * DD; i += 256) {
    int n = i >> 7, k = i & 127;
    wT[n * DD + k] = f2bf(w1[k * DD + n]);
  }
  __syncthreads();

  // GEMM1 + relu -> h
#pragma unroll
  for (int i = 0; i < 4; ++i) {
    int idx = i * 8 + wave, mt = idx >> 3, nt = idx & 7;
    f32x8 acc = {};
#pragma unroll
    for (int kk = 0; kk < DD; kk += 32)
      acc = wmma_bf16(load_frag(xA, mt * 16, DD, kk, lane),
                      load_frag(wT, nt * 16, DD, kk, lane), acc);
    int n = lane & 15, half = lane >> 4, col = nt * 16 + n;
    float bias = b1[col];
#pragma unroll
    for (int r = 0; r < 8; ++r) {
      float v = acc[r] + bias;
      hS[(mt * 16 + half * 8 + r) * DD + col] = f2bf(v > 0.f ? v : 0.f);
    }
  }
  __syncthreads();

  // residual GEMM
  for (int i = tid; i < DD * DD; i += 256) {
    int n = i >> 7, k = i & 127;
    wT[n * DD + k] = f2bf(wr[k * DD + n]);
  }
  __syncthreads();
  f32x8 racc[4];
#pragma unroll
  for (int i = 0; i < 4; ++i) {
    int idx = i * 8 + wave, mt = idx >> 3, nt = idx & 7;
    f32x8 acc = {};
#pragma unroll
    for (int kk = 0; kk < DD; kk += 32)
      acc = wmma_bf16(load_frag(xA, mt * 16, DD, kk, lane),
                      load_frag(wT, nt * 16, DD, kk, lane), acc);
    racc[i] = acc;
  }
  __syncthreads();

  // GEMM2 accumulate
  for (int i = tid; i < DD * DD; i += 256) {
    int n = i >> 7, k = i & 127;
    wT[n * DD + k] = f2bf(w2[k * DD + n]);
  }
  __syncthreads();
#pragma unroll
  for (int i = 0; i < 4; ++i) {
    int idx = i * 8 + wave, mt = idx >> 3, nt = idx & 7;
#pragma unroll
    for (int kk = 0; kk < DD; kk += 32)
      racc[i] = wmma_bf16(load_frag(hS, mt * 16, DD, kk, lane),
                          load_frag(wT, nt * 16, DD, kk, lane), racc[i]);
  }
  __syncthreads();

#pragma unroll
  for (int i = 0; i < 4; ++i) {
    int idx = i * 8 + wave, mt = idx >> 3, nt = idx & 7;
    int n = lane & 15, half = lane >> 4, col = nt * 16 + n;
    float bias = b2[col] + br[col];
#pragma unroll
    for (int r = 0; r < 8; ++r)
      yS[(mt * 16 + half * 8 + r) * DD + col] = racc[i][r] + bias;
  }
  __syncthreads();

  if (tid < 64) {
    const float* yr = yS + tid * DD;
    float mu = 0.f;
    for (int c = 0; c < DD; ++c) mu += yr[c];
    mu *= (1.0f / DD);
    float var = 0.f;
    for (int c = 0; c < DD; ++c) { float d = yr[c] - mu; var += d * d; }
    float rs = rsqrtf(var * (1.0f / DD) + EPSLN);
    size_t base = (size_t)(row0 + tid) * DD;
    for (int c = 0; c < DD; ++c)
      zout[base + c] = f2bf(g[c] * (yr[c] - mu) * rs + be[c]);
  }
}

// ---------------- kernel 3: MultiHeadEMA scan + SiLU ----------------
// in z: [BN][PNUM][D] bf16 ; out a: [BN][D][PNUM] bf16 (head K layout)
__global__ __launch_bounds__(256) void ema_kernel(
    const bf16_t* __restrict__ z,
    const float* __restrict__ delta, const float* __restrict__ alpha,
    const float* __restrict__ beta,  const float* __restrict__ gamma,
    const float* __restrict__ omega, bf16_t* __restrict__ aout) {
  int idx = blockIdx.x * 256 + threadIdx.x;   // BN*D threads
  int bn = idx >> 7, d = idx & 127;
  float p0 = sigmoidf_(delta[d * 2 + 0]), p1 = sigmoidf_(delta[d * 2 + 1]);
  float q0 = 1.0f - p0 * sigmoidf_(alpha[d * 2 + 0]);
  float q1 = 1.0f - p1 * sigmoidf_(alpha[d * 2 + 1]);
  float c0 = p0 * beta[d * 2 + 0], c1 = p1 * beta[d * 2 + 1];
  float g0 = gamma[d * 2 + 0] * EMASCALE, g1 = gamma[d * 2 + 1] * EMASCALE;
  float om = omega[d];
  const bf16_t* zr = z + (size_t)bn * PNUM * DD + d;
  bf16_t* outr = aout + (size_t)bn * KTOT + (size_t)d * PNUM;
  float s0 = 0.f, s1 = 0.f;
  for (int t = 0; t < PNUM; ++t) {
    float xv = bf2f(zr[(size_t)t * DD]);
    s0 = q0 * s0 + c0 * xv;
    s1 = q1 * s1 + c1 * xv;
    float o = s0 * g0 + s1 * g1 + xv * om;
    outr[t] = f2bf(o / (1.0f + __expf(-o)));   // silu
  }
}

// ---------------- kernel 4: flatten head GEMM [1344,32640]x[32640,96] ----------------
__global__ __launch_bounds__(256) void head_kernel(
    const bf16_t* __restrict__ a, const bf16_t* __restrict__ w,
    const float* __restrict__ hb, float* __restrict__ out) {
  __shared__ __align__(16) char smem[(32 * 128 + 96 * 128) * 2];  // 32KB
  bf16_t* aS = (bf16_t*)(smem);               // [32][128]
  bf16_t* bS = (bf16_t*)(smem + 32 * 128 * 2);// [96][128] (transposed)
  const int tid = threadIdx.x, lane = tid & 31, wave = tid >> 5;
  const int row0 = blockIdx.x * 32;

  int t0 = wave;          // tiles 0..7
  int t1 = wave + 8;      // tiles 8..11 on waves 0..3
  int mt0 = t0 / 6, nt0 = t0 % 6;
  int mt1 = t1 / 6, nt1 = t1 % 6;
  f32x8 acc0 = {}, acc1 = {};

  for (int k0 = 0; k0 < KTOT; k0 += 128) {
    __syncthreads();
    for (int i = tid; i < 512; i += 256) {     // A: 32 rows x 128 K (b128 loads)
      int r = i >> 4, q = i & 15;
      ((uint4*)aS)[i] = *(const uint4*)(a + (size_t)(row0 + r) * KTOT + k0 + q * 8);
    }
    for (int i = tid; i < 96 * 128; i += 256) { // B^T stage
      int n = i >> 7, k = i & 127;
      bS[n * 128 + k] = w[(size_t)(k0 + k) * PREDN + n];
    }
    // prefetch next K-chunk (global_prefetch_b8; fire-and-forget, no counter)
    if (k0 + 128 < KTOT) {
      if (tid < 32)
        __builtin_prefetch(a + (size_t)(row0 + tid) * KTOT + k0 + 128, 0, 0);
      if (tid < 128)
        __builtin_prefetch(w + (size_t)(k0 + 128 + tid) * PREDN, 0, 0);
    }
    __syncthreads();
#pragma unroll
    for (int kk = 0; kk < 128; kk += 32)
      acc0 = wmma_bf16(load_frag(aS, mt0 * 16, 128, kk, lane),
                       load_frag(bS, nt0 * 16, 128, kk, lane), acc0);
    if (wave < 4) {
#pragma unroll
      for (int kk = 0; kk < 128; kk += 32)
        acc1 = wmma_bf16(load_frag(aS, mt1 * 16, 128, kk, lane),
                         load_frag(bS, nt1 * 16, 128, kk, lane), acc1);
    }
  }

  int n = lane & 15, half = lane >> 4;
  {
    int col = nt0 * 16 + n;
    float bias = hb[col];
#pragma unroll
    for (int r = 0; r < 8; ++r)
      out[(size_t)(row0 + mt0 * 16 + half * 8 + r) * PREDN + col] = acc0[r] + bias;
  }
  if (wave < 4) {
    int col = nt1 * 16 + n;
    float bias = hb[col];
#pragma unroll
    for (int r = 0; r < 8; ++r)
      out[(size_t)(row0 + mt1 * 16 + half * 8 + r) * PREDN + col] = acc1[r] + bias;
  }
}

// ---------------- host launcher ----------------
extern "C" void kernel_launch(void* const* d_in, const int* in_sizes, int n_in,
                              void* d_out, int out_size, void* d_ws, size_t ws_size,
                              hipStream_t stream) {
  const float* x      = (const float*)d_in[0];
  const float* fp_w1  = (const float*)d_in[1];
  const float* fp_b1  = (const float*)d_in[2];
  const float* fp_w2  = (const float*)d_in[3];
  const float* fp_b2  = (const float*)d_in[4];
  const float* fp_wr  = (const float*)d_in[5];
  const float* fp_br  = (const float*)d_in[6];
  const float* fp_g   = (const float*)d_in[7];
  const float* fp_be  = (const float*)d_in[8];
  const float* enc_w1 = (const float*)d_in[9];
  const float* enc_b1 = (const float*)d_in[10];
  const float* enc_w2 = (const float*)d_in[11];
  const float* enc_b2 = (const float*)d_in[12];
  const float* enc_wr = (const float*)d_in[13];
  const float* enc_br = (const float*)d_in[14];
  const float* enc_g  = (const float*)d_in[15];
  const float* enc_be = (const float*)d_in[16];
  const float* e_del  = (const float*)d_in[17];
  const float* e_alp  = (const float*)d_in[18];
  const float* e_bet  = (const float*)d_in[19];
  const float* e_gam  = (const float*)d_in[20];
  const float* e_omg  = (const float*)d_in[21];
  const float* head_w = (const float*)d_in[22];
  const float* head_b = (const float*)d_in[23];
  float* out = (float*)d_out;

  const size_t ZBYTES = (size_t)NROWS * DD * sizeof(bf16_t);  // 87,736,320 B
  bf16_t* z0 = (bf16_t*)d_ws;
  bf16_t* z1 = (bf16_t*)((char*)d_ws + ZBYTES);
  bf16_t* hw = (bf16_t*)((char*)d_ws + 2 * ZBYTES);

  const int hwN = KTOT * PREDN;  // 3,133,440
  cvt_kernel<<<(hwN + 255) / 256, 256, 0, stream>>>(head_w, hw, hwN);

  fp_kernel<<<NROWS / 64, 256, 0, stream>>>(x, fp_w1, fp_b1, fp_w2, fp_b2,
                                            fp_wr, fp_br, fp_g, fp_be, z0);

  // encoder layer 0: z0 -> z1
  enc_kernel<<<NROWS / 64, 256, 0, stream>>>(z0, enc_w1, enc_b1, enc_w2, enc_b2,
                                             enc_wr, enc_br, enc_g, enc_be, z1);
  // encoder layer 1: z1 -> z0
  enc_kernel<<<NROWS / 64, 256, 0, stream>>>(z1, enc_w1 + DD * DD, enc_b1 + DD,
                                             enc_w2 + DD * DD, enc_b2 + DD,
                                             enc_wr + DD * DD, enc_br + DD,
                                             enc_g + DD, enc_be + DD, z0);

  // EMA scan: z0 -> z1 (relayout [BN][D][PNUM])
  ema_kernel<<<(BN * DD) / 256, 256, 0, stream>>>(z0, e_del, e_alp, e_bet,
                                                  e_gam, e_omg, z1);

  // head: z1 x hw -> out
  head_kernel<<<BN / 32, 256, 0, stream>>>(z1, hw, head_b, out);
}
